// PolicyHead_38053410243237
// MI455X (gfx1250) — matrix operand order repaired and checked by reading, not compiled, not run on hardware
//
#include <hip/hip_runtime.h>
#include <hip/hip_bf16.h>
#include <math.h>

typedef _Float16 f16;
typedef __attribute__((ext_vector_type(8)))  _Float16 v8h;
typedef __attribute__((ext_vector_type(16))) _Float16 v16h;
typedef __attribute__((ext_vector_type(8)))  float    v8f;

#define B_  128
#define T_  128
#define A_  16
#define P_  4
#define D_  256
#define H_  512
#define Q_  128
#define QI_ 512
#define V_  64

#define POS_PER_BLOCK 4
#define MROWS 64                 // POS_PER_BLOCK * A_
#define XSTRIDE (D_ + 16)        // halves, keeps 16B alignment + bank rotation
#define HSTRIDE (H_ + 16)        // halves
#define KSTRIDE (Q_ + 4)         // floats

// workspace offsets (bytes) for transposed f16/f32 weights
#define OFF_W1T 0
#define OFF_W2T (OFF_W1T + P_*D_*H_*2)
#define OFF_W3T (OFF_W2T + P_*H_*H_*2)
#define OFF_QWT (OFF_W3T + P_*H_*Q_*2)

// dynamic LDS layout (bytes)
#define SZ_X   (MROWS * XSTRIDE * 2)
#define SZ_H   (MROWS * HSTRIDE * 2)
#define SZ_KEY (MROWS * KSTRIDE * 4)
#define SZ_BB  (POS_PER_BLOCK * QI_ * 4)
#define SZ_QRY (POS_PER_BLOCK * Q_ * 4)
#define LDS_X    0
#define LDS_H1   (LDS_X   + SZ_X)
#define LDS_H2   (LDS_H1  + SZ_H)
#define LDS_KEY  (LDS_H2  + SZ_H)
#define LDS_BB   (LDS_KEY + SZ_KEY)
#define LDS_QRY  (LDS_BB  + SZ_BB)
#define LDS_INV  (LDS_QRY + SZ_QRY)
#define LDS_TOTAL (LDS_INV + MROWS * 4)   // 214,272 B < 320 KB/WGP

__device__ __forceinline__ v8f wmma16(v16h a, v16h b, v8f c) {
  // D = A(16x32 f16) x B(32x16 f16) + C(16x16 f32)
  return __builtin_amdgcn_wmma_f32_16x16x32_f16(
      false, a, false, b, (short)0, c, false, false);
}

// A fragment: LDS row-major [M][K], contiguous k order (same as load_b_frag):
// lane<16: M=lane, halves 0..15 <- K k0..k0+15
// lane>=16: M=lane-16, K base shifted by +16
__device__ __forceinline__ v16h load_a_frag(const f16* S, int stride, int m0, int k0, int lane) {
  const int m = m0 + (lane & 15);
  const int k = k0 + ((lane & 16) ? 16 : 0);
  const f16* p = S + m * stride + k;
  v8h lo = *(const v8h*)(p);
  v8h hi = *(const v8h*)(p + 8);
  return __builtin_shufflevector(lo, hi, 0,1,2,3,4,5,6,7,8,9,10,11,12,13,14,15);
}

// B fragment from transposed weight Wt[N][K] (f16, row-major, K contiguous):
// lane<16: N=lane, K = k0..k0+15 ; lane>=16: N=lane-16, K = k0+16..k0+31
__device__ __forceinline__ v16h load_b_frag(const f16* Wt, int ktot, int n0, int k0, int lane) {
  const int n = n0 + (lane & 15);
  const int k = k0 + ((lane & 16) ? 16 : 0);
  return *(const v16h*)(Wt + (size_t)(n) * ktot + k);   // 32B contiguous -> 2x b128
}

// 64 x 512 GEMM strip per block: wave owns 16x256 outputs (16 N-tiles),
// acc = 16 tiles x 8 VGPRs f32. Bias + ReLU epilogue, f16 store to LDS.
template<int KTOT>
__device__ __forceinline__ void gemm64x512_relu(
    const f16* __restrict__ Sin, int inStride,
    const f16* __restrict__ Wt, const float* __restrict__ bias,
    f16* __restrict__ Sout, int outStride, int wave, int lane)
{
  const int mt = (wave & 3) * 16;
  const int ntBase = (wave >> 2) * 16;
  const v8f vzero = {0.f,0.f,0.f,0.f,0.f,0.f,0.f,0.f};
  v8f acc[16];
  #pragma unroll
  for (int j = 0; j < 16; ++j) acc[j] = vzero;
  for (int s = 0; s < KTOT / 32; ++s) {
    v16h a = load_a_frag(Sin, inStride, mt, s * 32, lane);
    #pragma unroll
    for (int j = 0; j < 16; ++j) {
      v16h b = load_b_frag(Wt, KTOT, (ntBase + j) * 16, s * 32, lane);
      acc[j] = wmma16(a, b, acc[j]);
    }
  }
  const int nl = lane & 15;
  const int mAdd = (lane & 16) ? 8 : 0;   // C/D layout: m = v + (lane>=16 ? 8:0), n = lane&15
  #pragma unroll
  for (int j = 0; j < 16; ++j) {
    const int n = (ntBase + j) * 16 + nl;
    const float bv = bias[n];
    #pragma unroll
    for (int v = 0; v < 8; ++v) {
      float x = acc[j][v] + bv;
      x = fmaxf(x, 0.0f);
      Sout[(mt + v + mAdd) * outStride + n] = (f16)x;
    }
  }
}

// 64 x 128 GEMM (keyv): wave owns 16x64 outputs (4 N-tiles), f32 output + bias.
__device__ __forceinline__ void gemm64x128_key(
    const f16* __restrict__ Sin, int inStride,
    const f16* __restrict__ Wt, const float* __restrict__ bias,
    float* __restrict__ Sout, int outStride, int wave, int lane)
{
  const int mt = (wave & 3) * 16;
  const int ntBase = (wave >> 2) * 4;
  const v8f vzero = {0.f,0.f,0.f,0.f,0.f,0.f,0.f,0.f};
  v8f acc[4];
  #pragma unroll
  for (int j = 0; j < 4; ++j) acc[j] = vzero;
  for (int s = 0; s < H_ / 32; ++s) {
    v16h a = load_a_frag(Sin, inStride, mt, s * 32, lane);
    #pragma unroll
    for (int j = 0; j < 4; ++j) {
      v16h b = load_b_frag(Wt, H_, (ntBase + j) * 16, s * 32, lane);
      acc[j] = wmma16(a, b, acc[j]);
    }
  }
  const int nl = lane & 15;
  const int mAdd = (lane & 16) ? 8 : 0;
  #pragma unroll
  for (int j = 0; j < 4; ++j) {
    const int n = (ntBase + j) * 16 + nl;
    const float bv = bias[n];
    #pragma unroll
    for (int v = 0; v < 8; ++v)
      Sout[(mt + v + mAdd) * outStride + n] = acc[j][v] + bv;
  }
}

// ---- weight prep: [P][R][C] f32 -> [P][C][R] f16 / f32 ----
__global__ void transpose_f16_k(const float* __restrict__ src, f16* __restrict__ dst,
                                int np, int R, int C) {
  const int total = np * R * C;
  for (int i = blockIdx.x * blockDim.x + threadIdx.x; i < total; i += gridDim.x * blockDim.x) {
    const int p = i / (R * C);
    const int rem = i - p * (R * C);
    const int r = rem / C;
    const int c = rem - r * C;
    dst[(size_t)p * R * C + (size_t)c * R + r] = (f16)src[i];
  }
}
__global__ void transpose_f32_k(const float* __restrict__ src, float* __restrict__ dst,
                                int np, int R, int C) {
  const int total = np * R * C;
  for (int i = blockIdx.x * blockDim.x + threadIdx.x; i < total; i += gridDim.x * blockDim.x) {
    const int p = i / (R * C);
    const int rem = i - p * (R * C);
    const int r = rem / C;
    const int c = rem - r * C;
    dst[(size_t)p * R * C + (size_t)c * R + r] = src[i];
  }
}

// ---- fused policy head: embed+LN -> MLP(WMMA) -> attn -> log_softmax ----
__global__ void __launch_bounds__(256) policy_head_fused(
    const float* __restrict__ backbone, const int* __restrict__ valid,
    const int* __restrict__ phase_in, const int* __restrict__ trick,
    const float* __restrict__ emb0, const float* __restrict__ emb1,
    const float* __restrict__ emb2,
    const float* __restrict__ ln_g, const float* __restrict__ ln_b,
    const float* __restrict__ kb1, const float* __restrict__ kb2,
    const float* __restrict__ kb3, const float* __restrict__ qb,
    const f16* __restrict__ W1t, const f16* __restrict__ W2t,
    const f16* __restrict__ W3t, const float* __restrict__ qWt,
    float* __restrict__ out)
{
  extern __shared__ char smem[];
  f16*   sX   = (f16*)(smem + LDS_X);
  f16*   sH1  = (f16*)(smem + LDS_H1);
  f16*   sH2  = (f16*)(smem + LDS_H2);
  float* sKey = (float*)(smem + LDS_KEY);
  float* sBB  = (float*)(smem + LDS_BB);
  float* sQry = (float*)(smem + LDS_QRY);
  int*   sInv = (int*)(smem + LDS_INV);

  const int tid  = threadIdx.x;
  const int wave = tid >> 5;
  const int lane = tid & 31;
  const int pos0 = blockIdx.x * POS_PER_BLOCK;   // flat (b,t) index base
  const int t0   = pos0 % T_;
  const int ph   = t0 >> 5;                      // static phase routing: t*P/T

  // stage 0a: stage backbone rows in LDS
  for (int i = tid; i < POS_PER_BLOCK * QI_; i += 256)
    sBB[i] = backbone[(size_t)pos0 * QI_ + i];

  // stage 0b: action embeddings + LayerNorm -> sX (f16). One wave per row.
  for (int r = wave; r < MROWS; r += 8) {
    const int p = r >> 4, a = r & 15;
    const int pos = pos0 + p;
    const int base = (pos * A_ + a) * 3;
    const int raw0 = valid[base + 0];
    const int i0 = min(max(raw0, 0), V_ - 1);
    const int i1 = min(max(valid[base + 1], 0), V_ - 1);
    const int i2 = min(max(valid[base + 2], 0), V_ - 1);
    const int d0 = lane * 8;
    const float* e0 = emb0 + i0 * D_ + d0;
    const float* e1 = emb1 + i1 * D_ + d0;
    const float* e2 = emb2 + i2 * D_ + d0;
    float x[8];
    float s = 0.f, s2 = 0.f;
    #pragma unroll
    for (int u = 0; u < 8; ++u) {
      const float xv = e0[u] + e1[u] + e2[u];
      x[u] = xv; s += xv; s2 += xv * xv;
    }
    #pragma unroll
    for (int o = 16; o >= 1; o >>= 1) {
      s  += __shfl_xor(s,  o, 32);
      s2 += __shfl_xor(s2, o, 32);
    }
    const float mu   = s * (1.f / D_);
    const float var  = s2 * (1.f / D_) - mu * mu;
    const float rinv = rsqrtf(var + 1e-5f);
    #pragma unroll
    for (int u = 0; u < 8; ++u) {
      const float xn = (x[u] - mu) * rinv * ln_g[d0 + u] + ln_b[d0 + u];
      sX[r * XSTRIDE + d0 + u] = (f16)xn;
    }
    if (lane == 0) sInv[r] = (raw0 == -1) ? 1 : 0;
  }
  __syncthreads();

  // query: (4 x QI) @ qWt[ph] (Q x QI, transposed, f32) + qb  (small, VALU)
  for (int o = tid; o < POS_PER_BLOCK * Q_; o += 256) {
    const int p = o >> 7, q = o & (Q_ - 1);
    const float* w  = qWt + ((size_t)ph * Q_ + q) * QI_;
    const float* bb = sBB + p * QI_;
    float acc = qb[ph * Q_ + q];
    #pragma unroll 4
    for (int d = 0; d < QI_; ++d) acc += bb[d] * w[d];
    sQry[o] = acc;
  }

  // GEMM1: X(64x256) @ W1 -> H1(64x512), ReLU
  gemm64x512_relu<D_>(sX, XSTRIDE, W1t + (size_t)ph * H_ * D_, kb1 + ph * H_,
                      sH1, HSTRIDE, wave, lane);
  __syncthreads();
  // GEMM2: H1 @ W2 -> H2(64x512), ReLU
  gemm64x512_relu<H_>(sH1, HSTRIDE, W2t + (size_t)ph * H_ * H_, kb2 + ph * H_,
                      sH2, HSTRIDE, wave, lane);
  __syncthreads();
  // GEMM3: H2 @ W3 -> keyv(64x128) f32
  gemm64x128_key(sH2, HSTRIDE, W3t + (size_t)ph * Q_ * H_, kb3 + ph * Q_,
                 sKey, KSTRIDE, wave, lane);
  __syncthreads();

  // attention + masks + signal weight + log_softmax over A=16
  if (tid < MROWS) {
    const int r = tid, p = r >> 4, a = r & 15;
    const int pos = pos0 + p;
    float dot = 0.f;
    #pragma unroll 4
    for (int q = 0; q < Q_; ++q) dot += sQry[p * Q_ + q] * sKey[r * KSTRIDE + q];
    float attn = dot * 0.08838834764831845f;   // 1/sqrt(Q)

    int nv = 0;
    #pragma unroll
    for (int aa = 0; aa < A_; ++aa) nv += 1 - sInv[p * A_ + aa];

    if (a == 0 && phase_in[pos] == 1) {        // SIGNAL_PHASE
      const int tr = trick[pos];               // 0..12
      const float ps = 0.4f / (1.0f - __powf(0.6f, (float)(13 - tr)));
      float wgt = __logf(fmaxf((1.0f - ps) / ps * ((float)nv - 1.0f), 1e-5f));
      if (nv == 1) wgt = 0.f;
      attn += wgt;
    }
    if (sInv[r]) attn = -INFINITY;

    // half-wave (16-lane) reductions; wave32
    float m = attn;
    #pragma unroll
    for (int o = 8; o >= 1; o >>= 1) m = fmaxf(m, __shfl_xor(m, o, 16));
    const float e = __expf(attn - m);
    float ssum = e;
    #pragma unroll
    for (int o = 8; o >= 1; o >>= 1) ssum += __shfl_xor(ssum, o, 16);
    out[(size_t)pos * A_ + a] = attn - m - __logf(ssum);
  }
}

extern "C" void kernel_launch(void* const* d_in, const int* in_sizes, int n_in,
                              void* d_out, int out_size, void* d_ws, size_t ws_size,
                              hipStream_t stream) {
  (void)in_sizes; (void)n_in; (void)out_size; (void)ws_size;
  const float* backbone = (const float*)d_in[0];
  const int*   valid    = (const int*)d_in[1];
  const int*   phase    = (const int*)d_in[2];
  const int*   trick    = (const int*)d_in[3];
  const float* emb0     = (const float*)d_in[4];
  const float* emb1     = (const float*)d_in[5];
  const float* emb2     = (const float*)d_in[6];
  const float* ln_g     = (const float*)d_in[7];
  const float* ln_b     = (const float*)d_in[8];
  const float* kW1      = (const float*)d_in[9];
  const float* kb1      = (const float*)d_in[10];
  const float* kW2      = (const float*)d_in[11];
  const float* kb2      = (const float*)d_in[12];
  const float* kW3      = (const float*)d_in[13];
  const float* kb3      = (const float*)d_in[14];
  const float* qW       = (const float*)d_in[15];
  const float* qb       = (const float*)d_in[16];

  char* ws = (char*)d_ws;
  f16*   W1t = (f16*)(ws + OFF_W1T);   // [P][H][D] f16
  f16*   W2t = (f16*)(ws + OFF_W2T);   // [P][H][H] f16
  f16*   W3t = (f16*)(ws + OFF_W3T);   // [P][Q][H] f16
  float* qWt = (float*)(ws + OFF_QWT); // [P][Q][QI] f32

  transpose_f16_k<<<512, 256, 0, stream>>>(kW1, W1t, P_, D_, H_);
  transpose_f16_k<<<512, 256, 0, stream>>>(kW2, W2t, P_, H_, H_);
  transpose_f16_k<<<512, 256, 0, stream>>>(kW3, W3t, P_, H_, Q_);
  transpose_f32_k<<<512, 256, 0, stream>>>(qW,  qWt, P_, QI_, Q_);

  const dim3 grid(B_ * T_ / POS_PER_BLOCK);   // 4096 blocks, 1 WGP each
  policy_head_fused<<<grid, 256, LDS_TOTAL, stream>>>(
      backbone, valid, phase, trick, emb0, emb1, emb2, ln_g, ln_b,
      kb1, kb2, kb3, qb, W1t, W2t, W3t, qWt, (float*)d_out);
}
